// H3_Core_60653528154518
// MI455X (gfx1250) — matrix operand order repaired and checked by reading, not compiled, not run on hardware
//
#include <hip/hip_runtime.h>
#include <hip/hip_bf16.h>
#include <math.h>

typedef unsigned short ushort_t;
typedef unsigned int   uint_t;

// ---------------- problem constants ----------------
constexpr int kB = 512;    // batch
constexpr int kL = 256;    // clip tokens
constexpr int kH = 512;    // hidden
constexpr int kA = 128;    // attention dim
constexpr int kD = 512;    // IENC = VDIM = EDIM = CDIM
constexpr int kG = 4 * kH; // 2048 gates

// ---------------- WMMA fragment types ----------------
typedef __bf16 v16bf __attribute__((ext_vector_type(16)));
typedef float  v8f   __attribute__((ext_vector_type(8)));

union FragAB { v16bf v; uint_t u[8]; };
union FragC  { v8f  v; float f[8]; };
static_assert(sizeof(FragAB) == 32, "frag size");
static_assert(sizeof(FragC) == 32, "frag size");

// fp32 -> bf16 round-to-nearest-even, two packed per dword
__device__ __forceinline__ uint_t pack_bf2(float x0, float x1) {
    uint_t u0 = __float_as_uint(x0);
    uint_t u1 = __float_as_uint(x1);
    u0 += 0x7fffu + ((u0 >> 16) & 1u);
    u1 += 0x7fffu + ((u1 >> 16) & 1u);
    return (u0 >> 16) | (u1 & 0xffff0000u);
}

// A fragment (16x32 bf16): lane (g = lane>>4, hr = lane&15) holds row hr.
// u[0..3] = K in [g*8, g*8+7], u[4..7] = K in [16+g*8, 16+g*8+7]  (contiguous 16B chunks)
__device__ __forceinline__ void load_fragA_bf16(FragAB& f, const ushort_t* __restrict__ rowk, int g) {
    const uint4 q0 = *reinterpret_cast<const uint4*>(rowk + g * 8);
    const uint4 q1 = *reinterpret_cast<const uint4*>(rowk + 16 + g * 8);
    f.u[0] = q0.x; f.u[1] = q0.y; f.u[2] = q0.z; f.u[3] = q0.w;
    f.u[4] = q1.x; f.u[5] = q1.y; f.u[6] = q1.z; f.u[7] = q1.w;
}
// B fragment (32x16 bf16): lane holds column hr; u[0..3] = K in [g*8, g*8+7], u[4..7] = K in [16+g*8, 16+g*8+7]
__device__ __forceinline__ void load_fragB_bf16(FragAB& f, const ushort_t* __restrict__ rowk, int g) {
    const uint4 q0 = *reinterpret_cast<const uint4*>(rowk + g * 8);
    const uint4 q1 = *reinterpret_cast<const uint4*>(rowk + 16 + g * 8);
    f.u[0] = q0.x; f.u[1] = q0.y; f.u[2] = q0.z; f.u[3] = q0.w;
    f.u[4] = q1.x; f.u[5] = q1.y; f.u[6] = q1.z; f.u[7] = q1.w;
}

// ---------------- generic WMMA GEMM on pre-converted bf16 operands ----------------
// Out[M,N](f32) = A[M,K](bf16) @ W[N,K]^T(bf16) (+Cin) (+bias1+bias2)
__global__ void wmma_gemm_pre(const ushort_t* __restrict__ Ab, int lda,
                              const ushort_t* __restrict__ Wb, int ldw,
                              const float* __restrict__ Cin,
                              const float* __restrict__ bias1,
                              const float* __restrict__ bias2,
                              float* __restrict__ Out, int ldo,
                              int M, int N, int K)
{
    const int lane   = threadIdx.x & 31;
    const int wave   = threadIdx.x >> 5;
    const int nwaves = blockDim.x >> 5;
    const int tilesN = N >> 4;
    const int tilesM = M >> 4;
    const int tile   = blockIdx.x * nwaves + wave;
    if (tile >= tilesM * tilesN) return;          // wave-uniform: EXEC stays all-1s

    const int row0 = (tile / tilesN) << 4;
    const int col0 = (tile % tilesN) << 4;
    const int g  = lane >> 4;
    const int hr = lane & 15;

    FragC c;
    const int cn = col0 + hr;
    {
        float badd = 0.0f;
        if (bias1) badd += bias1[cn];
        if (bias2) badd += bias2[cn];
#pragma unroll
        for (int v = 0; v < 8; ++v) {
            const int cm = row0 + g * 8 + v;
            float cc = Cin ? Cin[(size_t)cm * ldo + cn] : 0.0f;
            c.f[v] = cc + badd;
        }
    }

    const ushort_t* arow = Ab + (size_t)(row0 + hr) * lda;
    const ushort_t* wrow = Wb + (size_t)(col0 + hr) * ldw;
    for (int k0 = 0; k0 < K; k0 += 32) {
        __builtin_prefetch(arow + k0 + 128, 0, 0);
        __builtin_prefetch(wrow + k0 + 128, 0, 0);
        FragAB a, b;
        load_fragA_bf16(a, arow + k0, g);
        load_fragB_bf16(b, wrow + k0, g);
        c.v = __builtin_amdgcn_wmma_f32_16x16x32_bf16(false, a.v, false, b.v,
                                                      (short)0, c.v, false, false);
    }
#pragma unroll
    for (int v = 0; v < 8; ++v)
        Out[(size_t)(row0 + g * 8 + v) * ldo + cn] = c.f[v];
}

// ---------------- fused Bahdanau attention: one workgroup per batch row ----------------
// scores[l] = wa . tanh(clip[b,l,:] @ wc^T + bc + att_h[b,:]) + ba  -> masked softmax
// att_res[b,:] = sum_l w[l] * clip[b,l,:]   (clip[b] = 512 KB, L2-resident between phases)
__global__ void attention_fused(const float* __restrict__ clip,
                                const int*   __restrict__ clip_mask,
                                const float* __restrict__ att_h,
                                const ushort_t* __restrict__ wcb,   // bf16 [A,D]
                                const float* __restrict__ bc,
                                const float* __restrict__ wa,
                                const float* __restrict__ ba,
                                float* __restrict__ att_res)
{
    constexpr int CH   = 32;        // clip rows staged per chunk
    constexpr int LROW = kD + 8;    // padded LDS row (ushorts): 1040B -> 4-bank lane stride
    __shared__ ushort_t clipb16[CH * LROW];   // ~33 KB
    __shared__ float sc[kL];
    __shared__ float ah[kA];
    __shared__ float tmp[256];

    const int b   = blockIdx.x;
    const int tid = threadIdx.x;           // 256 threads = 8 waves
    if (tid < kA) ah[tid] = att_h[(size_t)b * kA + tid];
    sc[tid] = 0.0f;

    const float* clipb = clip + (size_t)b * kL * kD;
    const int lane = tid & 31, wave = tid >> 5;
    const int g = lane >> 4, hr = lane & 15;

    for (int l0 = 0; l0 < kL; l0 += CH) {
        __syncthreads();
        // stage CH rows of clip[b] into LDS as bf16 (each element converted exactly once)
        for (int i = tid; i < CH * (kD / 2); i += 256) {
            const int r  = i / (kD / 2);
            const int cp = i % (kD / 2);
            const float* src = clipb + (size_t)(l0 + r) * kD + 2 * cp;
            *reinterpret_cast<uint_t*>(&clipb16[r * LROW + 2 * cp]) = pack_bf2(src[0], src[1]);
        }
        __syncthreads();

        // 2 row-tiles x 8 col-tiles per chunk; wave w owns column tile w
        for (int t = wave; t < 16; t += 8) {
            const int row0 = (t >> 3) << 4;   // 0 or 16
            const int col0 = (t & 7) << 4;
            FragC c;
#pragma unroll
            for (int v = 0; v < 8; ++v) c.f[v] = 0.0f;

            const ushort_t* arow = clipb16 + (row0 + hr) * LROW;
            const ushort_t* wrow = wcb + (size_t)(col0 + hr) * kD;
            for (int k0 = 0; k0 < kD; k0 += 32) {
                FragAB a, w;
                load_fragA_bf16(a, arow + k0, g);   // ds_load_b128 x2
                load_fragB_bf16(w, wrow + k0, g);   // global_load_b128 x2
                c.v = __builtin_amdgcn_wmma_f32_16x16x32_bf16(false, a.v, false, w.v,
                                                              (short)0, c.v, false, false);
            }
            const int n = col0 + hr;
            const float wan = wa[n];
            const float add = bc[n] + ah[n];
#pragma unroll
            for (int v = 0; v < 8; ++v) {
                const float s = tanhf(c.f[v] + add) * wan;
                atomicAdd(&sc[l0 + row0 + g * 8 + v], s);    // ds_add_f32
            }
        }
    }
    __syncthreads();

    // masked softmax over L (mask post-softmax + renorm == e*m / sum(e*m))
    const float s = sc[tid] + ba[0];
    tmp[tid] = s;
    __syncthreads();
    for (int off = 128; off > 0; off >>= 1) {
        if (tid < off) tmp[tid] = fmaxf(tmp[tid], tmp[tid + off]);
        __syncthreads();
    }
    const float mx = tmp[0];
    __syncthreads();
    const float e = expf(s - mx) * (float)clip_mask[(size_t)b * kL + tid];
    tmp[tid] = e;
    __syncthreads();
    for (int off = 128; off > 0; off >>= 1) {
        if (tid < off) tmp[tid] += tmp[tid + off];
        __syncthreads();
    }
    const float inv = 1.0f / tmp[0];
    sc[tid] = e * inv;
    __syncthreads();

    // weighted sum in fp32: each thread owns columns tid, tid+256 (coalesced, clip[b] in L2)
    float acc0 = 0.0f, acc1 = 0.0f;
    for (int l = 0; l < kL; ++l) {
        const float w = sc[l];
        const float* row = clipb + (size_t)l * kD;
        __builtin_prefetch(row + kD + tid, 0, 0);
        acc0 += w * row[tid];
        acc1 += w * row[tid + 256];
    }
    att_res[(size_t)b * kD + tid]       = acc0;
    att_res[(size_t)b * kD + tid + 256] = acc1;
}

// ---------------- elementwise LSTM cell (gates order i,f,g,o) ----------------
__device__ __forceinline__ float sigmoidf(float x) { return 1.0f / (1.0f + expf(-x)); }

__global__ void lstm_cell(const float* __restrict__ gates,   // [B, 4H]
                          const float* __restrict__ c_prev,  // [B, H]
                          float* __restrict__ h_out,
                          float* __restrict__ c_out,
                          float* __restrict__ h_out2)        // optional duplicate
{
    const int idx = blockIdx.x * blockDim.x + threadIdx.x;
    if (idx >= kB * kH) return;
    const int bb = idx / kH, j = idx % kH;
    const float* grow = gates + (size_t)bb * kG;
    const float i  = sigmoidf(grow[j]);
    const float f  = sigmoidf(grow[kH + j]);
    const float gg = tanhf(grow[2 * kH + j]);
    const float o  = sigmoidf(grow[3 * kH + j]);
    const float c  = f * c_prev[idx] + i * gg;
    const float h  = o * tanhf(c);
    c_out[idx] = c;
    h_out[idx] = h;
    if (h_out2) h_out2[idx] = h;
}

// ---------------- fp32 -> bf16 bulk convert (pairs) ----------------
__global__ void f32_to_bf16_pairs(const float* __restrict__ in, uint_t* __restrict__ out, int npairs) {
    const int i = blockIdx.x * blockDim.x + threadIdx.x;
    if (i >= npairs) return;
    out[i] = pack_bf2(in[2 * i], in[2 * i + 1]);
}

// ---------------- concat helpers, emitting bf16 pairs ----------------
__global__ void concat0_bf16(const float* __restrict__ xt,
                             const float* __restrict__ video,
                             const float* __restrict__ preh,
                             uint_t* __restrict__ x0)          // bf16 [B, 3D]
{
    const int i = blockIdx.x * blockDim.x + threadIdx.x;      // pair index
    if (i >= kB * (3 * kD) / 2) return;
    const int m  = i / ((3 * kD) / 2);
    const int j  = 2 * (i % ((3 * kD) / 2));
    float a, b;
    if (j < kD)          { const float* p = xt + (size_t)m * kD + j;              a = p[0]; b = p[1]; }
    else if (j < 2 * kD) { const float* p = video + (j - kD);                     a = p[0]; b = p[1]; }
    else                 { const float* p = preh + (size_t)m * kD + (j - 2 * kD); a = p[0]; b = p[1]; }
    x0[i] = pack_bf2(a, b);
}

__global__ void concat2_bf16(const float* __restrict__ s0,
                             const float* __restrict__ s1,
                             uint_t* __restrict__ xo)          // bf16 [B, 2D]
{
    const int i = blockIdx.x * blockDim.x + threadIdx.x;
    if (i >= kB * (2 * kD) / 2) return;
    const int m = i / kD;
    const int j = 2 * (i % kD);
    const float* p = (j < kD) ? (s0 + (size_t)m * kD + j) : (s1 + (size_t)m * kD + (j - kD));
    xo[i] = pack_bf2(p[0], p[1]);
}

// ---------------- host-side orchestration ----------------
extern "C" void kernel_launch(void* const* d_in, const int* in_sizes, int n_in,
                              void* d_out, int out_size, void* d_ws, size_t ws_size,
                              hipStream_t stream) {
    (void)in_sizes; (void)n_in; (void)out_size; (void)ws_size;
    const float* xt      = (const float*)d_in[0];
    const float* video   = (const float*)d_in[1];
    const float* event   = (const float*)d_in[2];
    const float* clip    = (const float*)d_in[3];
    const int*   clipm   = (const int*)  d_in[4];
    const float* state_h = (const float*)d_in[5];
    const float* state_c = (const float*)d_in[6];
    const float* w_ih0   = (const float*)d_in[7];
    const float* w_hh0   = (const float*)d_in[8];
    const float* b_ih0   = (const float*)d_in[9];
    const float* b_hh0   = (const float*)d_in[10];
    const float* w_ih1   = (const float*)d_in[11];
    const float* w_hh1   = (const float*)d_in[12];
    const float* b_ih1   = (const float*)d_in[13];
    const float* b_hh1   = (const float*)d_in[14];
    const float* w_ih2   = (const float*)d_in[15];
    const float* w_hh2   = (const float*)d_in[16];
    const float* b_ih2   = (const float*)d_in[17];
    const float* b_hh2   = (const float*)d_in[18];
    const float* wc      = (const float*)d_in[19];
    const float* bc      = (const float*)d_in[20];
    const float* wh      = (const float*)d_in[21];
    const float* bh      = (const float*)d_in[22];
    const float* wa      = (const float*)d_in[23];
    const float* ba      = (const float*)d_in[24];

    const size_t BH = (size_t)kB * kH;
    float* out  = (float*)d_out;
    float* h2a  = out;                 // h2 (first return value)
    float* newh = out + BH;            // new_h[3,B,H]
    float* newc = out + 4 * BH;        // new_c[3,B,H]
    float* h0 = newh, *h1 = newh + BH, *h2 = newh + 2 * BH;
    float* c0 = newc, *c1 = newc + BH, *c2 = newc + 2 * BH;

    // ---- workspace carve-up: bf16 region first, fp32 region after ----
    ushort_t* wsu = (ushort_t*)d_ws;
    size_t o = 0;
    auto alloc_u = [&](size_t n) { ushort_t* p = wsu + o; o += n; return p; };
    ushort_t* wih0b = alloc_u((size_t)kG * 3 * kD);
    ushort_t* whh0b = alloc_u((size_t)kG * kH);
    ushort_t* wih1b = alloc_u((size_t)kG * 2 * kD);
    ushort_t* whh1b = alloc_u((size_t)kG * kH);
    ushort_t* wih2b = alloc_u((size_t)kG * 2 * kD);
    ushort_t* whh2b = alloc_u((size_t)kG * kH);
    ushort_t* wcb   = alloc_u((size_t)kA * kD);
    ushort_t* whb   = alloc_u((size_t)kA * kD);
    ushort_t* shb   = alloc_u(3 * BH);          // state_h as bf16
    ushort_t* xb    = alloc_u((size_t)kB * 3 * kD);
    ushort_t* h1b   = alloc_u(BH);
    float* wsf    = (float*)(wsu + o);          // o is even -> 4B aligned
    float* gates  = wsf;                        // B*2048
    float* atth   = gates + (size_t)kB * kG;    // B*128
    float* attres = atth + (size_t)kB * kA;     // B*512

    const int TPB = 256;
    const int elemBlocks = (int)((BH + TPB - 1) / TPB);
    auto gemmBlocks = [](int M, int N) { return ((M / 16) * (N / 16) + 7) / 8; };
    auto conv = [&](const float* in, ushort_t* outp, size_t n) {
        const int npairs = (int)(n / 2);
        f32_to_bf16_pairs<<<(npairs + TPB - 1) / TPB, TPB, 0, stream>>>(in, (uint_t*)outp, npairs);
    };

    // ---- one-shot bf16 conversion of all reused operands ----
    conv(w_ih0, wih0b, (size_t)kG * 3 * kD);
    conv(w_hh0, whh0b, (size_t)kG * kH);
    conv(w_ih1, wih1b, (size_t)kG * 2 * kD);
    conv(w_hh1, whh1b, (size_t)kG * kH);
    conv(w_ih2, wih2b, (size_t)kG * 2 * kD);
    conv(w_hh2, whh2b, (size_t)kG * kH);
    conv(wc,    wcb,   (size_t)kA * kD);
    conv(wh,    whb,   (size_t)kA * kD);
    conv(state_h, shb, 3 * BH);

    // ---- layer 0: x0 = [xt | video | state_h[2]] ----
    concat0_bf16<<<(kB * 3 * kD / 2) / TPB, TPB, 0, stream>>>(xt, video, state_h + 2 * BH, (uint_t*)xb);
    wmma_gemm_pre<<<gemmBlocks(kB, kG), TPB, 0, stream>>>(
        xb, 3 * kD, wih0b, 3 * kD, nullptr, b_ih0, b_hh0, gates, kG, kB, kG, 3 * kD);
    wmma_gemm_pre<<<gemmBlocks(kB, kG), TPB, 0, stream>>>(
        shb, kH, whh0b, kH, gates, nullptr, nullptr, gates, kG, kB, kG, kH);
    lstm_cell<<<elemBlocks, TPB, 0, stream>>>(gates, state_c, h0, c0, nullptr);

    // ---- layer 1: x1 = [event | h0] ----
    concat2_bf16<<<(kB * 2 * kD / 2) / TPB, TPB, 0, stream>>>(event, h0, (uint_t*)xb);
    wmma_gemm_pre<<<gemmBlocks(kB, kG), TPB, 0, stream>>>(
        xb, 2 * kD, wih1b, 2 * kD, nullptr, b_ih1, b_hh1, gates, kG, kB, kG, 2 * kD);
    wmma_gemm_pre<<<gemmBlocks(kB, kG), TPB, 0, stream>>>(
        shb + BH, kH, whh1b, kH, gates, nullptr, nullptr, gates, kG, kB, kG, kH);
    lstm_cell<<<elemBlocks, TPB, 0, stream>>>(gates, state_c + BH, h1, c1, nullptr);

    // ---- attention: att_h = h1 @ wh^T + bh, then fused score/softmax/weighted-sum ----
    conv(h1, h1b, BH);
    wmma_gemm_pre<<<gemmBlocks(kB, kA), TPB, 0, stream>>>(
        h1b, kH, whb, kH, nullptr, bh, nullptr, atth, kA, kB, kA, kH);
    attention_fused<<<kB, TPB, 0, stream>>>(clip, clipm, atth, wcb, bc, wa, ba, attres);

    // ---- layer 2: x2 = [att_res | h1] ----
    concat2_bf16<<<(kB * 2 * kD / 2) / TPB, TPB, 0, stream>>>(attres, h1, (uint_t*)xb);
    wmma_gemm_pre<<<gemmBlocks(kB, kG), TPB, 0, stream>>>(
        xb, 2 * kD, wih2b, 2 * kD, nullptr, b_ih2, b_hh2, gates, kG, kB, kG, 2 * kD);
    wmma_gemm_pre<<<gemmBlocks(kB, kG), TPB, 0, stream>>>(
        shb + 2 * BH, kH, whh2b, kH, gates, nullptr, nullptr, gates, kG, kB, kG, kH);
    lstm_cell<<<elemBlocks, TPB, 0, stream>>>(gates, state_c + 2 * BH, h2, c2, h2a);
}
